// StyleGATLayer_44461501448526
// MI455X (gfx1250) — matrix operand order repaired and checked by reading, not compiled, not run on hardware
//
#include <hip/hip_runtime.h>
#include <hip/hip_bf16.h>
#include <hip/hip_fp16.h>

typedef __attribute__((ext_vector_type(16))) _Float16 v16h;
typedef __attribute__((ext_vector_type(8)))  float    v8f;

union Frag16 { uint4 u[2]; v16h h; };

#define B_  4
#define C_  2048
#define D_  256
#define H_  8
#define DK_ 32
#define FF_ 512

// ---------------------------------------------------------------- convert + transpose weights: W (K x N) f32 -> WT (N x K) f16
__global__ void cvt_transpose_kernel(const float* __restrict__ W,
                                     _Float16* __restrict__ WT, int K, int N) {
  int i = blockIdx.x * 256 + threadIdx.x;   // index into WT (coalesced writes)
  if (i < K * N) {
    int n = i / K, k = i - n * K;
    WT[i] = (_Float16)W[(size_t)k * N + n];
  }
}

// ---------------------------------------------------------------- layernorm (one wave per 256-elem row)
__global__ __launch_bounds__(256) void ln_kernel(const float* __restrict__ x,
                                                 const float* __restrict__ g,
                                                 const float* __restrict__ bt,
                                                 _Float16* __restrict__ out) {
  int w = threadIdx.x >> 5, lane = threadIdx.x & 31;
  size_t row = (size_t)blockIdx.x * 8 + w;
  const float* xr = x + row * D_;
  float v[8];
  float s = 0.f;
#pragma unroll
  for (int i = 0; i < 8; i++) { v[i] = xr[lane + i * 32]; s += v[i]; }
#pragma unroll
  for (int off = 16; off; off >>= 1) s += __shfl_xor(s, off, 32);
  float mu = s * (1.f / D_);
  float vs = 0.f;
#pragma unroll
  for (int i = 0; i < 8; i++) { float d = v[i] - mu; vs += d * d; }
#pragma unroll
  for (int off = 16; off; off >>= 1) vs += __shfl_xor(vs, off, 32);
  float rs = rsqrtf(vs * (1.f / D_) + 1e-5f);
  _Float16* orow = out + row * D_;
#pragma unroll
  for (int i = 0; i < 8; i++) {
    int col = lane + i * 32;
    orow[col] = (_Float16)((v[i] - mu) * rs * g[col] + bt[col]);
  }
}

// ---------------------------------------------------------------- WMMA GEMM, 64x64 tile, 8 waves, no LDS:
// A (M x K) row-major f16, WT (N x K) n-major f16; fragments load directly from global (L2-resident).
// modes: 0 = qkv scatter (q16,k16,vT16)   1/3 = resid + bias -> f32   2 = gelu -> f16
__global__ __launch_bounds__(256) void gemm_kernel(
    const _Float16* __restrict__ A, const _Float16* __restrict__ WT,
    const float* __restrict__ bias, const float* __restrict__ resid,
    void* __restrict__ out0, void* __restrict__ out1, void* __restrict__ out2,
    int K, int N, int mode) {
  int tid = threadIdx.x;
  int w = tid >> 5, lane = tid & 31;
  int hi = lane >> 4, l15 = lane & 15;
  int mrow = (w & 3) * 16, nwave = (w >> 2) * 32;
  int m0 = blockIdx.y * 64;
  int n0 = blockIdx.x * 64;

  const _Float16* Arow = A + (size_t)(m0 + mrow + l15) * K;
  const _Float16* Bn0  = WT + (size_t)(n0 + nwave + l15) * K;
  const _Float16* Bn1  = WT + (size_t)(n0 + nwave + 16 + l15) * K;
  int koffA = hi * 8;    // A frag: halves K = koffA..+7 and koffA+16..+23
  int koffB = hi * 8;    // B frag: halves K = koffB..+7 and koffB+16..+23

  v8f c0 = {}, c1 = {};
  for (int k0 = 0; k0 < K; k0 += 32) {
    Frag16 af, bf0, bf1;
    af.u[0]  = *(const uint4*)&Arow[k0 + koffA];
    af.u[1]  = *(const uint4*)&Arow[k0 + koffA + 16];
    bf0.u[0] = *(const uint4*)&Bn0[k0 + koffB];
    bf0.u[1] = *(const uint4*)&Bn0[k0 + koffB + 16];
    bf1.u[0] = *(const uint4*)&Bn1[k0 + koffB];
    bf1.u[1] = *(const uint4*)&Bn1[k0 + koffB + 16];
    c0 = __builtin_amdgcn_wmma_f32_16x16x32_f16(false, af.h, false, bf0.h, (short)0, c0, false, false);
    c1 = __builtin_amdgcn_wmma_f32_16x16x32_f16(false, af.h, false, bf1.h, (short)0, c1, false, false);
  }

#pragma unroll
  for (int nb = 0; nb < 2; nb++) {
    v8f cf = nb ? c1 : c0;
    int gn = n0 + nwave + nb * 16 + l15;
    float bv = bias[gn];
#pragma unroll
    for (int r = 0; r < 8; r++) {
      int gm = m0 + mrow + r + hi * 8;
      float val = cf[r] + bv;
      if (mode == 0) {
        int bb = gm >> 11, cc = gm & 2047;        // C_ = 2048
        int which = gn >> 8, rem = gn & 255;
        int hh = rem >> 5, dk = rem & 31;
        if (which == 0)
          ((_Float16*)out0)[((size_t)(bb * H_ + hh) * C_ + cc) * DK_ + dk] = (_Float16)val;
        else if (which == 1)
          ((_Float16*)out1)[((size_t)(bb * H_ + hh) * C_ + cc) * DK_ + dk] = (_Float16)val;
        else
          ((_Float16*)out2)[((size_t)(bb * H_ + hh) * DK_ + dk) * C_ + cc] = (_Float16)val;
      } else if (mode == 2) {
        float ge = 0.5f * val * (1.f + erff(val * 0.70710678118654752f));
        ((_Float16*)out0)[(size_t)gm * N + gn] = (_Float16)ge;
      } else {  // modes 1 and 3: residual add, f32 out
        size_t gi = (size_t)gm * N + gn;
        ((float*)out0)[gi] = resid[gi] + val;
      }
    }
  }
}

// ---------------------------------------------------------------- fused flash attention, 4 waves, 16 q-rows/wave
__global__ __launch_bounds__(128) void attn_kernel(
    const _Float16* __restrict__ q16, const _Float16* __restrict__ k16,
    const _Float16* __restrict__ vT16, const float* __restrict__ adj,
    _Float16* __restrict__ attn16) {
  __shared__ __align__(16) _Float16 Pl[4][16 * 32];
  int tid = threadIdx.x;
  int w = tid >> 5, lane = tid & 31, hi = lane >> 4, l15 = lane & 15;
  int bh = blockIdx.y;
  int b = bh >> 3, hh = bh & 7;
  int qb = blockIdx.x * 64 + w * 16;

  const _Float16* qp = q16 + (size_t)bh * C_ * DK_;
  const _Float16* kp = k16 + (size_t)bh * C_ * DK_;
  const _Float16* vp = vT16 + (size_t)bh * DK_ * C_;
  const float* ap = adj + ((size_t)b * C_ + qb) * C_;

  Frag16 qa;
  {
    int m = qb + l15, koff = hi * 8;
    qa.u[0] = *(const uint4*)&qp[(size_t)m * DK_ + koff];
    qa.u[1] = *(const uint4*)&qp[(size_t)m * DK_ + koff + 16];
  }
  v8f o0 = {}, o1 = {};
  float mrow[8], lrow[8];
#pragma unroll
  for (int r = 0; r < 8; r++) { mrow[r] = -3.0e38f; lrow[r] = 0.f; }
  const float scale = 0.17677669529663687f;  // 1/sqrt(DK)
  _Float16* P = Pl[w];

  for (int kc = 0; kc < C_; kc += 32) {
    Frag16 kb0, kb1;
    {
      int n = kc + l15, koff = hi * 8;
      kb0.u[0] = *(const uint4*)&kp[(size_t)n * DK_ + koff];
      kb0.u[1] = *(const uint4*)&kp[(size_t)n * DK_ + koff + 16];
      kb1.u[0] = *(const uint4*)&kp[(size_t)(n + 16) * DK_ + koff];
      kb1.u[1] = *(const uint4*)&kp[(size_t)(n + 16) * DK_ + koff + 16];
    }
    v8f z = {};
    v8f s0 = __builtin_amdgcn_wmma_f32_16x16x32_f16(false, qa.h, false, kb0.h, (short)0, z, false, false);
    v8f s1 = __builtin_amdgcn_wmma_f32_16x16x32_f16(false, qa.h, false, kb1.h, (short)0, z, false, false);

#pragma unroll
    for (int r = 0; r < 8; r++) {
      int qr = r + hi * 8;
      const float* arow = ap + (size_t)qr * C_ + kc;
      if (kc + 32 < C_) __builtin_prefetch(arow + 32 + l15, 0, 0);
      s0[r] = s0[r] * scale + arow[l15];
      s1[r] = s1[r] * scale + arow[16 + l15];
    }
#pragma unroll
    for (int r = 0; r < 8; r++) {
      float t = fmaxf(s0[r], s1[r]);
#pragma unroll
      for (int off = 8; off; off >>= 1) t = fmaxf(t, __shfl_xor(t, off, 32));
      float mn = fmaxf(mrow[r], t);
      float corr = __expf(mrow[r] - mn);
      float p0 = __expf(s0[r] - mn);
      float p1 = __expf(s1[r] - mn);
      float rs = p0 + p1;
#pragma unroll
      for (int off = 8; off; off >>= 1) rs += __shfl_xor(rs, off, 32);
      lrow[r] = lrow[r] * corr + rs;
      mrow[r] = mn;
      o0[r] *= corr;
      o1[r] *= corr;
      int prow = r + hi * 8;
      P[prow * 32 + l15] = (_Float16)p0;        // C/D layout -> row-major P tile
      P[prow * 32 + 16 + l15] = (_Float16)p1;
    }
    Frag16 pa, vb0, vb1;
    {
      int koff = hi * 8;                         // re-read P in A-fragment layout
      pa.u[0] = *(const uint4*)&P[l15 * 32 + koff];
      pa.u[1] = *(const uint4*)&P[l15 * 32 + koff + 16];
    }
    {
      int koff = hi * 8;
      vb0.u[0] = *(const uint4*)&vp[(size_t)l15 * C_ + kc + koff];
      vb0.u[1] = *(const uint4*)&vp[(size_t)l15 * C_ + kc + koff + 16];
      vb1.u[0] = *(const uint4*)&vp[(size_t)(16 + l15) * C_ + kc + koff];
      vb1.u[1] = *(const uint4*)&vp[(size_t)(16 + l15) * C_ + kc + koff + 16];
    }
    o0 = __builtin_amdgcn_wmma_f32_16x16x32_f16(false, pa.h, false, vb0.h, (short)0, o0, false, false);
    o1 = __builtin_amdgcn_wmma_f32_16x16x32_f16(false, pa.h, false, vb1.h, (short)0, o1, false, false);
  }

#pragma unroll
  for (int r = 0; r < 8; r++) {
    float inv = 1.f / lrow[r];
    int qr = qb + r + hi * 8;
    size_t base = ((size_t)b * C_ + qr) * D_ + hh * DK_;
    attn16[base + l15] = (_Float16)(o0[r] * inv);
    attn16[base + 16 + l15] = (_Float16)(o1[r] * inv);
  }
}

// ---------------------------------------------------------------- launcher
extern "C" void kernel_launch(void* const* d_in, const int* in_sizes, int n_in,
                              void* d_out, int out_size, void* d_ws, size_t ws_size,
                              hipStream_t stream) {
  const float* h    = (const float*)d_in[0];
  const float* adj  = (const float*)d_in[1];
  const float* Wqkv = (const float*)d_in[2];
  const float* bqkv = (const float*)d_in[3];
  const float* Wout = (const float*)d_in[4];
  const float* bout = (const float*)d_in[5];
  const float* g1   = (const float*)d_in[6];
  const float* be1  = (const float*)d_in[7];
  const float* g2   = (const float*)d_in[8];
  const float* be2  = (const float*)d_in[9];
  const float* Wf1  = (const float*)d_in[10];
  const float* bf1  = (const float*)d_in[11];
  const float* Wf2  = (const float*)d_in[12];
  const float* bf2  = (const float*)d_in[13];
  float* out = (float*)d_out;

  const size_t BC = (size_t)B_ * C_;  // 8192 rows
  char* ws = (char*)d_ws;
  size_t off = 0;
  auto alloc = [&](size_t bytes) {
    void* p = ws + off;
    off += (bytes + 255) & ~(size_t)255;
    return p;
  };
  _Float16* hn16   = (_Float16*)alloc(BC * D_ * 2);
  _Float16* q16    = (_Float16*)alloc(BC * D_ * 2);
  _Float16* k16    = (_Float16*)alloc(BC * D_ * 2);
  _Float16* vT16   = (_Float16*)alloc(BC * D_ * 2);
  _Float16* attn16 = (_Float16*)alloc(BC * D_ * 2);
  _Float16* h2_16  = (_Float16*)alloc(BC * D_ * 2);
  _Float16* mid16  = (_Float16*)alloc(BC * FF_ * 2);
  _Float16* wq16   = (_Float16*)alloc((size_t)D_ * 3 * D_ * 2);   // (3D x D) n-major
  _Float16* wo16   = (_Float16*)alloc((size_t)D_ * D_ * 2);      // (D x D) n-major
  _Float16* wf116  = (_Float16*)alloc((size_t)D_ * FF_ * 2);     // (FF x D) n-major
  _Float16* wf216  = (_Float16*)alloc((size_t)FF_ * D_ * 2);     // (D x FF) n-major
  float* hmid      = (float*)alloc(BC * D_ * 4);

  // weight convert + transpose to n-major f16
  cvt_transpose_kernel<<<(D_ * 3 * D_ + 255) / 256, 256, 0, stream>>>(Wqkv, wq16, D_, 3 * D_);
  cvt_transpose_kernel<<<(D_ * D_ + 255) / 256, 256, 0, stream>>>(Wout, wo16, D_, D_);
  cvt_transpose_kernel<<<(D_ * FF_ + 255) / 256, 256, 0, stream>>>(Wf1, wf116, D_, FF_);
  cvt_transpose_kernel<<<(FF_ * D_ + 255) / 256, 256, 0, stream>>>(Wf2, wf216, FF_, D_);

  // LN1
  ln_kernel<<<BC / 8, 256, 0, stream>>>(h, g1, be1, hn16);

  // QKV GEMM: (8192 x 256) @ (256 x 768), scatter into q/k/vT f16
  gemm_kernel<<<dim3(12, 128), 256, 0, stream>>>(hn16, wq16, bqkv, nullptr,
                                                 q16, k16, vT16, D_, 3 * D_, 0);

  // fused attention
  attn_kernel<<<dim3(C_ / 64, B_ * H_), 128, 0, stream>>>(q16, k16, vT16, adj, attn16);

  // out-proj + residual -> hmid (f32)
  gemm_kernel<<<dim3(4, 128), 256, 0, stream>>>(attn16, wo16, bout, h,
                                                hmid, nullptr, nullptr, D_, D_, 1);

  // LN2
  ln_kernel<<<BC / 8, 256, 0, stream>>>(hmid, g2, be2, h2_16);

  // FFN1 + gelu -> mid16 (f16)
  gemm_kernel<<<dim3(8, 128), 256, 0, stream>>>(h2_16, wf116, bf1, nullptr,
                                                mid16, nullptr, nullptr, D_, FF_, 2);

  // FFN2 + residual -> d_out (f32)
  gemm_kernel<<<dim3(4, 128), 256, 0, stream>>>(mid16, wf216, bf2, hmid,
                                                out, nullptr, nullptr, FF_, D_, 3);
}
